// ODEBlock_76776835383580
// MI455X (gfx1250) — matrix operand order repaired and checked by reading, not compiled, not run on hardware
//
#include <hip/hip_runtime.h>

typedef __attribute__((ext_vector_type(16))) __bf16 v16bf;
typedef __attribute__((ext_vector_type(8)))  float  v8f;
typedef unsigned int   u32;
typedef unsigned short u16;

#define B_TOT    4096
#define F_A      64
#define F_S      256
#define HID      1024
#define T_STEPS  64
#define K1       (F_A + F_S)   /* 320 */
#define BM       32
#define NTHREADS 256

/* LDS row strides in dwords, padded so column walks are bank-conflict free */
#define SROW 161   /* s tile: 320 bf16 = 160 dw, +1 pad  */
#define HROW 513   /* h tile: 1024 bf16 = 512 dw, +1 pad */
#define YROW 258   /* y tile: 256 f32, +2 pad (keeps float2 alignment) */
#define SMEM_DW   (BM*SROW + BM*HROW + BM*YROW)
#define SMEM_BYTES (SMEM_DW * 4)

__device__ __forceinline__ u16 f32_to_bf16(float x) {
  u32 u = __builtin_bit_cast(u32, x);
  u32 r = u + 0x7FFFu + ((u >> 16) & 1u);   /* round-to-nearest-even */
  return (u16)(r >> 16);
}

__device__ __forceinline__ u32 pack2_bf16(float lo, float hi) {
  return (u32)f32_to_bf16(lo) | ((u32)f32_to_bf16(hi) << 16);
}

/* Hardware tanh if the toolchain exposes it (gfx1250 V_TANH_F32); otherwise a
 * branchless exp2/rcp sequence — never the divergent libm path.             */
__device__ __forceinline__ float fast_tanh(float x) {
#if __has_builtin(__builtin_amdgcn_tanhf)
  return __builtin_amdgcn_tanhf(x);
#elif __has_builtin(__builtin_amdgcn_tanh_f32)
  return __builtin_amdgcn_tanh_f32(x);
#else
  float ax = __builtin_fabsf(x);
  /* e^{-2ax} = 2^{-2*log2(e)*ax} */
  float t = __builtin_amdgcn_exp2f(-2.885390081777927f * ax);
  float r = (1.0f - t) * __builtin_amdgcn_rcpf(1.0f + t);
  return __builtin_copysignf(r, x);
#endif
}

/* ---- Pack a KxN fp32 weight matrix into bf16 WMMA B-fragment order. ----
 * Fragment (kt, nt): 32 lanes x 8 dwords, n = lane&15, dword j = bf16 pair
 * k = (j>>2)*16 + (lane>>4)*8 + (j&3)*2 (+0,+1) within the 32-wide K chunk.
 * Packed dword index: ((kt*NT + nt)*32 + lane)*8 + j                        */
__global__ void pack_w_bf16(const float* __restrict__ W, u32* __restrict__ P,
                            int N, int NT, int nFrag) {
  int d = blockIdx.x * blockDim.x + threadIdx.x;
  if (d >= nFrag * 256) return;
  int frag = d >> 8;
  int r    = d & 255;
  int lane = r >> 3;
  int j    = r & 7;
  int kt   = frag / NT;
  int nt   = frag - kt * NT;
  int k    = kt * 32 + (j >> 2) * 16 + (lane >> 4) * 8 + (j & 3) * 2;
  int n    = nt * 16 + (lane & 15);
  P[d] = pack2_bf16(W[(size_t)k * N + n], W[(size_t)(k + 1) * N + n]);
}

/* Load one 16x32 bf16 A fragment from an LDS tile (row-major, rowStride in
 * dwords).  Per ISA 7.12.2: lane half selects K octet, vgpr v: pair p=v&3,
 * group g=v>>2 -> k = kbase + g*16 + half*8 + p*2 (+0,+1 packed).           */
__device__ __forceinline__ v16bf load_a_frag(const u32* base, int rowStride,
                                             int mt, int kbase, int lane) {
  const int half = lane >> 4;
  const int row  = (lane & 15) + mt * 16;
  const u32* rp  = base + row * rowStride;
  union { u32 u[8]; v16bf v; } f;
#pragma unroll
  for (int v = 0; v < 8; ++v) {
    int p = v & 3, g = v >> 2;
    int k = kbase + g * 16 + half * 8 + p * 2;
    f.u[v] = rp[k >> 1];
  }
  return f.v;
}

/* Load one 32x16 bf16 B fragment from the packed weight array (contiguous
 * 32B per lane -> two b128 loads).                                          */
__device__ __forceinline__ v16bf load_b_frag(const u32* __restrict__ pack,
                                             int NT, int kt, int nt, int lane) {
  const uint4* p4 = (const uint4*)(pack + (((kt * NT + nt) << 5) + lane) * 8);
  uint4 x0 = p4[0];
  uint4 x1 = p4[1];
  union { u32 u[8]; v16bf v; } f;
  f.u[0] = x0.x; f.u[1] = x0.y; f.u[2] = x0.z; f.u[3] = x0.w;
  f.u[4] = x1.x; f.u[5] = x1.y; f.u[6] = x1.z; f.u[7] = x1.w;
  return f.v;
}

__global__ __launch_bounds__(NTHREADS)
void ode_scan_kernel(const float* __restrict__ actions,
                     const float* __restrict__ y0,
                     const float* __restrict__ b1,
                     const float* __restrict__ b2,
                     const u32*  __restrict__ packW1,
                     const u32*  __restrict__ packW2,
                     const float* __restrict__ dtp,
                     float* __restrict__ out) {
  extern __shared__ char smem[];
  u32*  sDw = (u32*)smem;                /* s tile, bf16, BM x SROW dwords  */
  u32*  hDw = sDw + BM * SROW;           /* h tile, bf16, BM x HROW dwords  */
  float* yT = (float*)(hDw + BM * HROW); /* y state fp32 BM x YROW          */
  u16*  hH  = (u16*)hDw;

  const int tid  = threadIdx.x;
  const int lane = tid & 31;
  const int w    = tid >> 5;
  const int b0   = blockIdx.x * BM;
  const float dt = *dtp;

  const int mt   = w & 1;        /* which 16-row half of the batch tile */
  const int wq   = w >> 1;       /* 0..3 : N-range owner                */
  const int half = lane >> 4;
  const int col  = lane & 15;

  /* init state from y0 and emit trajectory at t = 0 */
  for (int i = tid; i < BM * F_S; i += NTHREADS) {
    int m = i >> 8, n = i & 255;
    float v = y0[(b0 + m) * F_S + n];
    yT[m * YROW + n] = v;
    out[(size_t)((b0 + m) * F_S + n) * T_STEPS] = v;
  }
  __syncthreads();

  for (int t = 0; t < T_STEPS - 1; ++t) {
    /* ---- stage s = [a_t | y] as bf16 pairs (full-dword DS stores) ---- */
    for (int i = tid; i < BM * (F_A / 2); i += NTHREADS) {
      int m  = i >> 5;             /* /32 pairs per row */
      int f2 = (i & 31) * 2;
      const float* ap = actions + ((size_t)(b0 + m) * F_A + f2) * T_STEPS + t;
      sDw[m * SROW + (f2 >> 1)] = pack2_bf16(ap[0], ap[T_STEPS]);
    }
    for (int i = tid; i < BM * (F_S / 2); i += NTHREADS) {
      int m  = i >> 7;             /* /128 pairs per row */
      int n2 = (i & 127) * 2;
      float2 yv = *(const float2*)(yT + m * YROW + n2);
      sDw[m * SROW + (F_A / 2) + (n2 >> 1)] = pack2_bf16(yv.x, yv.y);
    }
    __syncthreads();

    /* ---- GEMM1: h = tanh(s @ W1 + b1), each wave: 16 tiles in groups of 4 */
    for (int g4 = 0; g4 < 4; ++g4) {
      const int nb = wq * 16 + g4 * 4;
      v8f acc[4];
#pragma unroll
      for (int j = 0; j < 4; ++j)
#pragma unroll
        for (int r = 0; r < 8; ++r) acc[j][r] = 0.0f;

      for (int kt = 0; kt < K1 / 32; ++kt) {          /* 10 K steps */
        v16bf a = load_a_frag(sDw, SROW, mt, kt * 32, lane);
#pragma unroll
        for (int j = 0; j < 4; ++j) {
          v16bf b = load_b_frag(packW1, HID / 16, kt, nb + j, lane);
          acc[j] = __builtin_amdgcn_wmma_f32_16x16x32_bf16(
              false, a, false, b, (short)0, acc[j], false, false);
        }
      }
#pragma unroll
      for (int j = 0; j < 4; ++j) {
        int n = (nb + j) * 16 + col;
        float bias = b1[n];
#pragma unroll
        for (int r = 0; r < 8; ++r) {
          int m = mt * 16 + r + half * 8;             /* C/D lane layout */
          float hv = fast_tanh(acc[j][r] + bias);
          hH[m * (2 * HROW) + n] = f32_to_bf16(hv);
        }
      }
    }
    __syncthreads();

    /* ---- GEMM2: y += dt*(h @ W2 + b2); each wave: 4 tiles ---- */
    {
      const int nb = wq * 4;
      v8f acc[4];
#pragma unroll
      for (int j = 0; j < 4; ++j)
#pragma unroll
        for (int r = 0; r < 8; ++r) acc[j][r] = 0.0f;

      for (int kt = 0; kt < HID / 32; ++kt) {         /* 32 K steps */
        v16bf a = load_a_frag(hDw, HROW, mt, kt * 32, lane);
#pragma unroll
        for (int j = 0; j < 4; ++j) {
          v16bf b = load_b_frag(packW2, F_S / 16, kt, nb + j, lane);
          acc[j] = __builtin_amdgcn_wmma_f32_16x16x32_bf16(
              false, a, false, b, (short)0, acc[j], false, false);
        }
      }
#pragma unroll
      for (int j = 0; j < 4; ++j) {
        int n = (nb + j) * 16 + col;
        float bias = b2[n];
#pragma unroll
        for (int r = 0; r < 8; ++r) {
          int m = mt * 16 + r + half * 8;
          float yo = yT[m * YROW + n];
          float yn = yo + dt * (acc[j][r] + bias);
          yT[m * YROW + n] = yn;
          out[(size_t)((b0 + m) * F_S + n) * T_STEPS + (t + 1)] = yn;
        }
      }
    }
    __syncthreads();   /* y fully updated before next step stages s */
  }
}

extern "C" void kernel_launch(void* const* d_in, const int* in_sizes, int n_in,
                              void* d_out, int out_size, void* d_ws, size_t ws_size,
                              hipStream_t stream) {
  /* inputs: 0 ts, 1 actions, 2 y0, 3 W1, 4 b1, 5 W2, 6 b2, 7 dt */
  const float* actions = (const float*)d_in[1];
  const float* y0      = (const float*)d_in[2];
  const float* W1      = (const float*)d_in[3];
  const float* b1      = (const float*)d_in[4];
  const float* W2      = (const float*)d_in[5];
  const float* b2      = (const float*)d_in[6];
  const float* dtp     = (const float*)d_in[7];
  float* out = (float*)d_out;

  /* workspace: packed bf16 weights in WMMA fragment order */
  u32* packW1 = (u32*)d_ws;                 /* 320*1024/2 = 163840 dwords */
  u32* packW2 = packW1 + (K1 * HID / 2);    /* 1024*256/2 = 131072 dwords */

  const int frags1 = (K1 / 32) * (HID / 16);   /* 640 */
  const int frags2 = (HID / 32) * (F_S / 16);  /* 512 */
  pack_w_bf16<<<frags1, 256, 0, stream>>>(W1, packW1, HID, HID / 16, frags1);
  pack_w_bf16<<<frags2, 256, 0, stream>>>(W2, packW2, F_S, F_S / 16, frags2);

  ode_scan_kernel<<<B_TOT / BM, NTHREADS, SMEM_BYTES, stream>>>(
      actions, y0, b1, b2, packW1, packW2, dtp, out);
}
